// AutoRegistrationLayer_64269890617431
// MI455X (gfx1250) — matrix operand order repaired and checked by reading, not compiled, not run on hardware
//
#include <hip/hip_runtime.h>

typedef __attribute__((ext_vector_type(2)))  _Float16 v2h;
typedef __attribute__((ext_vector_type(8)))  _Float16 v8h;
typedef __attribute__((ext_vector_type(16))) _Float16 v16h;
typedef __attribute__((ext_vector_type(8)))  float    v8f;

#define NODES 50000
#define NEDGE 1600000

__device__ __forceinline__ float celu1(float v) {
  return v > 0.f ? v : expm1f(v);
}

__device__ __forceinline__ v16h cat16(v8h lo, v8h hi) {
  return __builtin_shufflevector(lo, hi, 0,1,2,3,4,5,6,7,8,9,10,11,12,13,14,15);
}

// B fragment for v_wmma_f32_16x16x32_f16 from a transposed (N-major, K-contiguous)
// f16 weight matrix Bt[64][64]. Covers K = ks*32 .. +31, N = nb*16 .. +15.
// Layout (wave32): lanes 0-15 hold K={0..7,16..23} for N=lane, lanes 16-31 hold K={8..15,24..31}.
__device__ __forceinline__ v16h load_bfrag(const _Float16* __restrict__ bt,
                                           int lane, int nb, int ks) {
  int n = nb * 16 + (lane & 15);
  int kbase = ks * 32 + ((lane & 31) < 16 ? 0 : 8);
  const _Float16* p = bt + n * 64 + kbase;
  v8h lo = *(const v8h*)(p);
  v8h hi = *(const v8h*)(p + 16);
  return cat16(lo, hi);
}

// A fragment from an LDS row-major f16 tile h[rows][64].
// Lanes 0-15: M=lane, K = {0..7, 16..23}; lanes 16-31: M=lane-16, K = {8..15, 24..31}.
__device__ __forceinline__ v16h load_afrag(const _Float16* __restrict__ sh64,
                                           int lane, int mrow, int ks) {
  int m = lane & 15;
  int k0 = ks * 32 + ((lane & 31) < 16 ? 0 : 8);
  const _Float16* p = sh64 + (mrow + m) * 64 + k0;
  v8h lo = *(const v8h*)(p);        // K = k0 .. k0+7
  v8h hi = *(const v8h*)(p + 16);   // K = k0+16 .. k0+23
  return cat16(lo, hi);
}

__global__ void k_zero(float4* p, int n4) {
  int i = blockIdx.x * blockDim.x + threadIdx.x;
  if (i < n4) p[i] = make_float4(0.f, 0.f, 0.f, 0.f);
}

// Convert the three 64x64 f32 weight blocks to f16, transposed to N-major.
__global__ void k_prep(const float* __restrict__ w2, const float* __restrict__ g1,
                       const float* __restrict__ g2,
                       _Float16* __restrict__ w2t, _Float16* __restrict__ g1t,
                       _Float16* __restrict__ g2t) {
  for (int i = threadIdx.x; i < 64 * 64; i += 256) {
    int k = i >> 6, n = i & 63;
    w2t[n * 64 + k] = (_Float16)w2[i];
    g1t[n * 64 + k] = (_Float16)g1[i];   // g_w1 rows 0..63 of [67][64]
    g2t[n * 64 + k] = (_Float16)g2[i];
  }
}

// Edge phase: layer1 (VALU, K=6) -> LDS f16 -> layer2 (WMMA) -> relu -> atomic scatter-max.
__global__ __launch_bounds__(256)
void k_edge(const float* __restrict__ x, const float* __restrict__ pos,
            const int* __restrict__ ei, const float* __restrict__ w1,
            const float* __restrict__ b1, const _Float16* __restrict__ w2t,
            const float* __restrict__ b2, float* __restrict__ aggr) {
  __shared__ _Float16 sh[256 * 64];   // h tile, 32 KB
  __shared__ float sw1[6 * 64];
  __shared__ float sb1[64];
  __shared__ float sb2[64];
  __shared__ int   sdst[256];

  const int t = threadIdx.x;
  const int e0 = blockIdx.x * 256;

  if (t < 64) { sb1[t] = b1[t]; sb2[t] = b2[t]; }
  for (int i = t; i < 6 * 64; i += 256) sw1[i] = w1[i];

  // Gather per-edge inputs (x/pos are L2-resident: 600 KB each).
  int e = e0 + t;
  float m0 = 0.f, m1 = 0.f, m2 = 0.f, m3 = 0.f, m4 = 0.f, m5 = 0.f;
  int d = -1;
  if (e < NEDGE) {
    int s = ei[e];
    d = ei[NEDGE + e];
    m0 = x[s * 3 + 0]; m1 = x[s * 3 + 1]; m2 = x[s * 3 + 2];
    m3 = pos[s * 3 + 0] - pos[d * 3 + 0];
    m4 = pos[s * 3 + 1] - pos[d * 3 + 1];
    m5 = pos[s * 3 + 2] - pos[d * 3 + 2];
  }
  sdst[t] = d;
  __syncthreads();

  // Layer 1: h = celu(m @ W1 + b1), stored f16 row-major into LDS.
  #pragma unroll 4
  for (int f = 0; f < 64; f += 2) {
    float a0 = sb1[f]     + m0 * sw1[f]       + m1 * sw1[64 + f]
             + m2 * sw1[128 + f]   + m3 * sw1[192 + f]
             + m4 * sw1[256 + f]   + m5 * sw1[320 + f];
    float a1 = sb1[f + 1] + m0 * sw1[f + 1]   + m1 * sw1[64 + f + 1]
             + m2 * sw1[128 + f + 1] + m3 * sw1[192 + f + 1]
             + m4 * sw1[256 + f + 1] + m5 * sw1[320 + f + 1];
    v2h hv = { (_Float16)celu1(a0), (_Float16)celu1(a1) };
    *(v2h*)&sh[t * 64 + f] = hv;
  }
  __syncthreads();

  // Layer 2 via WMMA: each wave owns 2 M-tiles (16 edges each), all 64 outputs.
  const int wave = t >> 5, lane = t & 31;
  v16h bf[4][2];
  #pragma unroll
  for (int nb = 0; nb < 4; ++nb)
    #pragma unroll
    for (int ks = 0; ks < 2; ++ks)
      bf[nb][ks] = load_bfrag(w2t, lane, nb, ks);

  const int half16 = (lane < 16) ? 0 : 8;
  const int feat = lane & 15;

  // Per-lane biases for the 4 N-tiles, hoisted out of the hot loops.
  float bias[4];
  #pragma unroll
  for (int nb = 0; nb < 4; ++nb) bias[nb] = sb2[nb * 16 + feat];

  #pragma unroll
  for (int mt = 0; mt < 2; ++mt) {
    int mrow = (wave * 2 + mt) * 16;
    v16h a0 = load_afrag(sh, lane, mrow, 0);
    v16h a1 = load_afrag(sh, lane, mrow, 1);
    v8f c[4];
    #pragma unroll
    for (int nb = 0; nb < 4; ++nb) {
      c[nb] = (v8f){};
      c[nb] = __builtin_amdgcn_wmma_f32_16x16x32_f16(false, a0, false, bf[nb][0],
                                                     (short)0, c[nb], false, false);
      c[nb] = __builtin_amdgcn_wmma_f32_16x16x32_f16(false, a1, false, bf[nb][1],
                                                     (short)0, c[nb], false, false);
    }
    // relu + scatter-max, j-outer: the 4 N-tile outputs of a given j share one
    // aggr row -> one sdst read, one row pointer, constant 64B offsets on the
    // atomics. msg >= 0 and aggr init 0 => int-max on bits == float-max; zero
    // messages are skipped entirely.
    #pragma unroll
    for (int j = 0; j < 8; ++j) {
      int row = mrow + j + half16;
      int dd = sdst[row];
      if (dd >= 0) {
        int* rowp = (int*)aggr + dd * 64 + feat;
        #pragma unroll
        for (int nb = 0; nb < 4; ++nb) {
          float v = c[nb][j] + bias[nb];
          if (v > 0.f)
            atomicMax(rowp + nb * 16, __float_as_int(v));
        }
      }
    }
  }
}

// Node phase: u = celu([aggr,x] @ g_w1 + g_b1); out = celu(u @ g_w2 + g_b2).
__global__ __launch_bounds__(256)
void k_node(const float* __restrict__ x, const float* __restrict__ aggr,
            const _Float16* __restrict__ g1t, const float* __restrict__ g1,
            const float* __restrict__ gb1, const _Float16* __restrict__ g2t,
            const float* __restrict__ gb2, float* __restrict__ out) {
  __shared__ _Float16 sa[128 * 64];   // aggr tile f16, 16 KB
  __shared__ _Float16 su[128 * 64];   // u tile f16, 16 KB
  __shared__ float sx[128 * 3];
  __shared__ float swx[3 * 64];       // g_w1 rows 64..66
  __shared__ float sgb1[64];
  __shared__ float sgb2[64];

  const int t = threadIdx.x;
  const int n0 = blockIdx.x * 128;

  if (t < 64) { sgb1[t] = gb1[t]; sgb2[t] = gb2[t]; }
  if (t < 192) swx[t] = g1[64 * 64 + t];
  for (int i = t; i < 128 * 3; i += 256) {
    int node = n0 + i / 3;
    sx[i] = (node < NODES) ? x[node * 3 + (i % 3)] : 0.f;
  }
  for (int i = t; i < 128 * 64; i += 256) {
    int node = n0 + (i >> 6);
    sa[i] = (node < NODES) ? (_Float16)aggr[node * 64 + (i & 63)] : (_Float16)0.f;
  }
  __syncthreads();

  const int wave = t >> 5, lane = t & 31;
  const int mrow = wave * 16;
  const int half16 = (lane < 16) ? 0 : 8;
  const int feat = lane & 15;

  // Layer 1 (K=64 WMMA + rank-3 x-correction).
  {
    v16h a0 = load_afrag(sa, lane, mrow, 0);
    v16h a1 = load_afrag(sa, lane, mrow, 1);
    v8f c[4];
    #pragma unroll
    for (int nb = 0; nb < 4; ++nb) {
      v16h b0 = load_bfrag(g1t, lane, nb, 0);
      v16h b1v = load_bfrag(g1t, lane, nb, 1);
      c[nb] = (v8f){};
      c[nb] = __builtin_amdgcn_wmma_f32_16x16x32_f16(false, a0, false, b0,
                                                     (short)0, c[nb], false, false);
      c[nb] = __builtin_amdgcn_wmma_f32_16x16x32_f16(false, a1, false, b1v,
                                                     (short)0, c[nb], false, false);
    }
    float w0[4], w1v[4], w2v[4], bb[4];
    #pragma unroll
    for (int nb = 0; nb < 4; ++nb) {
      int n = nb * 16 + feat;
      w0[nb] = swx[n]; w1v[nb] = swx[64 + n]; w2v[nb] = swx[128 + n];
      bb[nb] = sgb1[n];
    }
    #pragma unroll
    for (int j = 0; j < 8; ++j) {
      int row = mrow + j + half16;
      float x0 = sx[row * 3 + 0], x1 = sx[row * 3 + 1], x2 = sx[row * 3 + 2];
      _Float16* up = &su[row * 64 + feat];
      #pragma unroll
      for (int nb = 0; nb < 4; ++nb) {
        float v = c[nb][j] + bb[nb] + x0 * w0[nb] + x1 * w1v[nb] + x2 * w2v[nb];
        up[nb * 16] = (_Float16)celu1(v);
      }
    }
  }
  __syncthreads();

  // Layer 2.
  {
    v16h a0 = load_afrag(su, lane, mrow, 0);
    v16h a1 = load_afrag(su, lane, mrow, 1);
    v8f c[4];
    #pragma unroll
    for (int nb = 0; nb < 4; ++nb) {
      v16h b0 = load_bfrag(g2t, lane, nb, 0);
      v16h b1v = load_bfrag(g2t, lane, nb, 1);
      c[nb] = (v8f){};
      c[nb] = __builtin_amdgcn_wmma_f32_16x16x32_f16(false, a0, false, b0,
                                                     (short)0, c[nb], false, false);
      c[nb] = __builtin_amdgcn_wmma_f32_16x16x32_f16(false, a1, false, b1v,
                                                     (short)0, c[nb], false, false);
    }
    float bb[4];
    #pragma unroll
    for (int nb = 0; nb < 4; ++nb) bb[nb] = sgb2[nb * 16 + feat];
    #pragma unroll
    for (int j = 0; j < 8; ++j) {
      int node = n0 + mrow + j + half16;
      if (node < NODES) {
        float* op = out + node * 64 + feat;
        #pragma unroll
        for (int nb = 0; nb < 4; ++nb)
          op[nb * 16] = celu1(c[nb][j] + bb[nb]);
      }
    }
  }
}

extern "C" void kernel_launch(void* const* d_in, const int* in_sizes, int n_in,
                              void* d_out, int out_size, void* d_ws, size_t ws_size,
                              hipStream_t stream) {
  (void)in_sizes; (void)n_in; (void)out_size; (void)ws_size;
  const float* x   = (const float*)d_in[0];
  const float* pos = (const float*)d_in[1];
  const int*   ei  = (const int*)d_in[2];
  const float* fw1 = (const float*)d_in[3];
  const float* fb1 = (const float*)d_in[4];
  const float* fw2 = (const float*)d_in[5];
  const float* fb2 = (const float*)d_in[6];
  const float* gw1 = (const float*)d_in[7];
  const float* gb1 = (const float*)d_in[8];
  const float* gw2 = (const float*)d_in[9];
  const float* gb2 = (const float*)d_in[10];
  float* out = (float*)d_out;

  char* ws = (char*)d_ws;
  float*    aggr = (float*)ws;                                   // N*64 f32 = 12.8 MB
  _Float16* w2t  = (_Float16*)(ws + (size_t)NODES * 64 * 4);     // 8 KB, 16B-aligned
  _Float16* g1t  = w2t + 64 * 64;
  _Float16* g2t  = g1t + 64 * 64;

  const int n4 = NODES * 64 / 4;
  k_zero<<<(n4 + 255) / 256, 256, 0, stream>>>((float4*)aggr, n4);
  k_prep<<<1, 256, 0, stream>>>(fw2, gw1, gw2, w2t, g1t, g2t);
  k_edge<<<(NEDGE + 255) / 256, 256, 0, stream>>>(x, pos, ei, fw1, fb1, w2t, fb2, aggr);
  k_node<<<(NODES + 127) / 128, 256, 0, stream>>>(x, aggr, g1t, gw1, gb1, g2t, gb2, out);
}
